// Model_1803886264686
// MI455X (gfx1250) — compile-verified, matrix operand order checked by reading, not run on hardware
//
#include <hip/hip_runtime.h>
#include <hip/hip_bf16.h>
#include <math.h>

// ---------------------------------------------------------------------------
// 2-layer tanh RNN (B=512, T=512, H=512) + per-step FC (V=128), fully fused.
// One persistent workgroup per 16-row batch tile (recurrence is independent
// per batch element -> no cross-WG sync). bf16 WMMA w/ fp32 accumulation.
// Depth-1 software pipeline on A (LDS) and B (L2) fragments; NT logits stores
// keep the weights resident in L2 across all 512 steps.
// ---------------------------------------------------------------------------

#define HID   512
#define BATCH 512
#define TLEN  512
#define VOC   128

typedef __bf16 v16bf __attribute__((ext_vector_type(16)));
typedef __bf16 v8bf  __attribute__((ext_vector_type(8)));
typedef float  v8f   __attribute__((ext_vector_type(8)));

// workspace layout (bf16 elements from base)
#define WS_WHH0 0
#define WS_WIH1 (512*512)
#define WS_WHH1 (2*512*512)
#define WS_FCW  (3*512*512)
#define WS_BF16_ELEMS (3*512*512 + 128*512)          // 851968
#define WS_F32_BYTE_OFF (WS_BF16_ELEMS * 2)          // 1703936 (16B aligned)

// ---------------------------------------------------------------------------
// prep: fp32 -> bf16 weight conversion + fused bias vectors
// ---------------------------------------------------------------------------
__global__ void prep_kernel(const float* __restrict__ whh0,
                            const float* __restrict__ wih1,
                            const float* __restrict__ whh1,
                            const float* __restrict__ fcw,
                            const float* __restrict__ bih0,
                            const float* __restrict__ bhh0,
                            const float* __restrict__ bih1,
                            const float* __restrict__ bhh1,
                            __bf16* __restrict__ wsb,
                            float*  __restrict__ wsf) {
  int i = blockIdx.x * blockDim.x + threadIdx.x;     // 0 .. 262143
  wsb[WS_WHH0 + i] = (__bf16)whh0[i];
  wsb[WS_WIH1 + i] = (__bf16)wih1[i];
  wsb[WS_WHH1 + i] = (__bf16)whh1[i];
  if (i < VOC * HID) wsb[WS_FCW + i] = (__bf16)fcw[i];
  if (i < HID) {
    wsf[i]       = bih0[i] + bhh0[i];
    wsf[HID + i] = bih1[i] + bhh1[i];
  }
}

// ---------------------------------------------------------------------------
// fragment load helpers
// ---------------------------------------------------------------------------
__device__ __forceinline__ v16bf load_frag(const __bf16* base, int kc) {
  union { v16bf v; v8bf h[2]; } u;
  u.h[0] = *(const v8bf*)(base + kc);
  u.h[1] = *(const v8bf*)(base + kc + 16);
  return u.v;
}

__device__ __forceinline__ v8f wmma_bf16(v16bf a, v16bf b, v8f c) {
  return __builtin_amdgcn_wmma_f32_16x16x32_bf16(false, a, false, b,
                                                 (short)0, c, false, false);
}

// 4 N-tiles per wave, depth-1 software pipeline over the K dimension:
// fragments for chunk k+1 are issued before the 4 WMMAs of chunk k.
__device__ __forceinline__ void gemm4(const __bf16* __restrict__ abase,
                                      const __bf16* const* __restrict__ brow,
                                      v8f* acc) {
  v16bf a_c = load_frag(abase, 0);
  v16bf b_c[4];
#pragma unroll
  for (int i = 0; i < 4; ++i) b_c[i] = load_frag(brow[i], 0);

#pragma unroll
  for (int kc = 0; kc < HID - 32; kc += 32) {
    v16bf a_n = load_frag(abase, kc + 32);     // ds_load_b128 x2 (next)
    v16bf b_n[4];
#pragma unroll
    for (int i = 0; i < 4; ++i)
      b_n[i] = load_frag(brow[i], kc + 32);    // global_load_b128 x8 (next)
#pragma unroll
    for (int i = 0; i < 4; ++i)
      acc[i] = wmma_bf16(a_c, b_c[i], acc[i]); // overlaps with next loads
    a_c = a_n;
#pragma unroll
    for (int i = 0; i < 4; ++i) b_c[i] = b_n[i];
  }
#pragma unroll
  for (int i = 0; i < 4; ++i)
    acc[i] = wmma_bf16(a_c, b_c[i], acc[i]);   // last chunk
}

// single N-tile version (FC head), same pipelining
__device__ __forceinline__ void gemm1(const __bf16* __restrict__ abase,
                                      const __bf16* __restrict__ brow,
                                      v8f& acc) {
  v16bf a_c = load_frag(abase, 0);
  v16bf b_c = load_frag(brow, 0);
#pragma unroll
  for (int kc = 0; kc < HID - 32; kc += 32) {
    v16bf a_n = load_frag(abase, kc + 32);
    v16bf b_n = load_frag(brow, kc + 32);
    acc = wmma_bf16(a_c, b_c, acc);
    a_c = a_n;
    b_c = b_n;
  }
  acc = wmma_bf16(a_c, b_c, acc);
}

// ---------------------------------------------------------------------------
// main fused persistent kernel: 32 WGs x 256 threads (8 wave32)
// ---------------------------------------------------------------------------
__global__ __launch_bounds__(256, 1)
void rnn_fused_kernel(const float* __restrict__ x,      // [B, 1, T]
                      const float* __restrict__ wih0,   // [H] (W_ih0[:,0])
                      const float* __restrict__ fcb,    // [V]
                      const __bf16* __restrict__ wsb,
                      const float* __restrict__ wsf,
                      float* __restrict__ out) {
  __shared__ __align__(16) __bf16 h0s[16 * HID];
  __shared__ __align__(16) __bf16 h1s[16 * HID];
  __shared__ float xs[16];

  const int tid   = threadIdx.x;
  const int wave  = tid >> 5;
  const int lane  = tid & 31;
  const int ncol  = lane & 15;
  const int half  = lane >> 4;       // 0 | 1
  const int khalf = half * 8;        // K offset inside fragment
  const int mrow  = half * 8;        // C-fragment M offset
  const int m_a   = lane & 15;       // A-fragment M (row of activation tile)
  const int mbase = blockIdx.x * 16;

  // zero-init activation tiles
  for (int i = tid; i < 16 * HID; i += 256) {
    h0s[i] = (__bf16)0.f;
    h1s[i] = (__bf16)0.f;
  }

  // per-lane constants: 4 N-tiles (64 cols) per wave for the H=512 GEMMs
  int nglob[4];
  const __bf16 *whh0r[4], *wih1r[4], *whh1r[4];
  float b0r[4], b1r[4], w0r[4];
#pragma unroll
  for (int i = 0; i < 4; ++i) {
    const int ng = (wave * 4 + i) * 16 + ncol;
    nglob[i] = ng;
    whh0r[i] = wsb + WS_WHH0 + (size_t)ng * HID + khalf;
    wih1r[i] = wsb + WS_WIH1 + (size_t)ng * HID + khalf;
    whh1r[i] = wsb + WS_WHH1 + (size_t)ng * HID + khalf;
    b0r[i] = wsf[ng];
    b1r[i] = wsf[HID + ng];
    w0r[i] = wih0[ng];
  }
  // FC head: one 16-col tile per wave (8 waves * 16 = V=128)
  const int nfc = wave * 16 + ncol;
  const __bf16* fcwr = wsb + WS_FCW + (size_t)nfc * HID + khalf;
  const float   fcbr = fcb[nfc];

  const __bf16* a0 = h0s + m_a * HID + khalf;
  const __bf16* a1 = h1s + m_a * HID + khalf;

  float* outh0 = out + (size_t)BATCH * TLEN * VOC;
  float* outh1 = outh0 + (size_t)BATCH * HID;

  for (int t = 0; t < TLEN; ++t) {
    __syncthreads();
    if (tid < 16) xs[tid] = __builtin_nontemporal_load(
                                &x[(size_t)(mbase + tid) * TLEN + t]);
    __syncthreads();
    const bool last = (t == TLEN - 1);

    // ------- layer 0: h0 = tanh(x_t*w_ih0 + h0 @ W_hh0^T + biases) -------
    v8f acc[4] = {};
    gemm4(a0, whh0r, acc);
    __bf16 hn[4][8];
#pragma unroll
    for (int i = 0; i < 4; ++i) {
#pragma unroll
      for (int r = 0; r < 8; ++r) {
        float v = acc[i][r] + xs[mrow + r] * w0r[i] + b0r[i];
        v = tanhf(v);
        if (last) outh0[(size_t)(mbase + mrow + r) * HID + nglob[i]] = v;
        hn[i][r] = (__bf16)v;
      }
    }
    __syncthreads();                       // everyone done reading old h0
#pragma unroll
    for (int i = 0; i < 4; ++i)
#pragma unroll
      for (int r = 0; r < 8; ++r)
        h0s[(mrow + r) * HID + nglob[i]] = hn[i][r];
    __syncthreads();

    // ------- layer 1: h1 = tanh(h0 @ W_ih1^T + h1 @ W_hh1^T + biases) -------
    v8f acc1[4] = {};
    gemm4(a0, wih1r, acc1);                // new h0
    gemm4(a1, whh1r, acc1);                // old h1
#pragma unroll
    for (int i = 0; i < 4; ++i) {
#pragma unroll
      for (int r = 0; r < 8; ++r) {
        float v = acc1[i][r] + b1r[i];
        v = tanhf(v);
        if (last) outh1[(size_t)(mbase + mrow + r) * HID + nglob[i]] = v;
        hn[i][r] = (__bf16)v;
      }
    }
    __syncthreads();
#pragma unroll
    for (int i = 0; i < 4; ++i)
#pragma unroll
      for (int r = 0; r < 8; ++r)
        h1s[(mrow + r) * HID + nglob[i]] = hn[i][r];
    __syncthreads();

    // ------- FC head: logits[b, t, :] = h1 @ fc_W^T + fc_b -------
    // Anti-hoist barrier on a 32-bit OFFSET (not the pointer): keeps the
    // compiler from caching the whole 256-bf16/lane FC row in 128 VGPRs
    // across the t-loop, while preserving the pointer's global-addrspace
    // provenance so these stay global_load_b128 (LOADcnt only), not
    // flat_load_b128 (LOADcnt+DScnt coupling + 64-bit VGPR address).
    int zoff = 0;
    asm volatile("" : "+v"(zoff));
    const __bf16* fcw_t = fcwr + zoff;

    v8f af = {};
    gemm1(a1, fcw_t, af);
#pragma unroll
    for (int r = 0; r < 8; ++r) {
      __builtin_nontemporal_store(
          af[r] + fcbr,
          &out[(size_t)(mbase + mrow + r) * TLEN * VOC + (size_t)t * VOC + nfc]);
    }
  }
}

// ---------------------------------------------------------------------------
extern "C" void kernel_launch(void* const* d_in, const int* in_sizes, int n_in,
                              void* d_out, int out_size, void* d_ws, size_t ws_size,
                              hipStream_t stream) {
  const float* x    = (const float*)d_in[0];
  const float* Wih0 = (const float*)d_in[1];
  const float* Whh0 = (const float*)d_in[2];
  const float* bih0 = (const float*)d_in[3];
  const float* bhh0 = (const float*)d_in[4];
  const float* Wih1 = (const float*)d_in[5];
  const float* Whh1 = (const float*)d_in[6];
  const float* bih1 = (const float*)d_in[7];
  const float* bhh1 = (const float*)d_in[8];
  const float* fcW  = (const float*)d_in[9];
  const float* fcb  = (const float*)d_in[10];

  __bf16* wsb = (__bf16*)d_ws;
  float*  wsf = (float*)((char*)d_ws + WS_F32_BYTE_OFF);

  // 262144 threads: one per element of each 512x512 matrix
  prep_kernel<<<1024, 256, 0, stream>>>(Whh0, Wih1, Whh1, fcW,
                                        bih0, bhh0, bih1, bhh1, wsb, wsf);

  // 32 batch tiles of 16 rows; whole recurrence fused in one launch
  rnn_fused_kernel<<<32, 256, 0, stream>>>(x, Wih0, fcb, wsb, wsf,
                                           (float*)d_out);
}
